// SwinTiny_10161892622367
// MI455X (gfx1250) — matrix operand order repaired and checked by reading, not compiled, not run on hardware
//
#include <hip/hip_runtime.h>

// ---------------------------------------------------------------------------
// Swin-Tiny forward for gfx1250 (MI455X). All GEMMs + attention on
// v_wmma_f32_16x16x32_bf16 (fp32 accumulate); GEMM tiles staged into LDS with
// global_load_async_to_lds_b128 (ASYNCcnt); elementwise glue on VALU.
// ---------------------------------------------------------------------------

typedef __bf16 bf16_t;
typedef __attribute__((ext_vector_type(16))) __bf16 bf16x16;
typedef __attribute__((ext_vector_type(8)))  float  f32x8;

union Frag16 { bf16x16 v; uint4 u[2]; };

__device__ __forceinline__ f32x8 wmma_bf16(const Frag16& a, const Frag16& b, f32x8 c) {
  return __builtin_amdgcn_wmma_f32_16x16x32_bf16(false, a.v, false, b.v,
                                                 (short)0, c, false, false);
}

// Async copy 64B (one 32-elem bf16 row) global -> LDS. Per-lane; EXEC-predicated.
__device__ __forceinline__ void async_cp_row64(unsigned lds, unsigned long long ga) {
  asm volatile(
      "global_load_async_to_lds_b128 %0, %1, off\n\t"
      "global_load_async_to_lds_b128 %0, %1, off offset:16\n\t"
      "global_load_async_to_lds_b128 %0, %1, off offset:32\n\t"
      "global_load_async_to_lds_b128 %0, %1, off offset:48"
      :: "v"(lds), "v"(ga) : "memory");
}
// Async copy 32B global -> LDS.
__device__ __forceinline__ void async_cp_half32(unsigned lds, unsigned long long ga) {
  asm volatile(
      "global_load_async_to_lds_b128 %0, %1, off\n\t"
      "global_load_async_to_lds_b128 %0, %1, off offset:16"
      :: "v"(lds), "v"(ga) : "memory");
}
__device__ __forceinline__ void wait_async0() {
  asm volatile("s_wait_asynccnt 0x0" ::: "memory");
}

// ---------------------------------------------------------------------------
// fp32 -> bf16 convert (activations & weights)
// ---------------------------------------------------------------------------
__global__ void k_f2bf(const float* __restrict__ in, bf16_t* __restrict__ out,
                       size_t n) {
  size_t i = (size_t)blockIdx.x * blockDim.x + threadIdx.x;
  if (i < n) out[i] = (bf16_t)in[i];
}

// ---------------------------------------------------------------------------
// Patch embedding: 4x4/4 conv as 48-dot per (token, out-channel). NCHW input.
// out: (16,56,56,96) token-major fp32
// ---------------------------------------------------------------------------
__global__ void k_patch_embed(const float* __restrict__ x,
                              const float* __restrict__ w,
                              const float* __restrict__ b,
                              float* __restrict__ out) {
  int idx = blockIdx.x * blockDim.x + threadIdx.x;
  const int total = 16 * 56 * 56 * 96;
  if (idx >= total) return;
  int oc = idx % 96;
  int t  = idx / 96;
  int ow = t % 56; t /= 56;
  int oh = t % 56;
  int bn = t / 56;
  float acc = b[oc];
  for (int c = 0; c < 3; ++c)
    for (int dy = 0; dy < 4; ++dy)
      for (int dx = 0; dx < 4; ++dx)
        acc += x[(((size_t)bn * 3 + c) * 224 + (oh * 4 + dy)) * 224 + (ow * 4 + dx)]
             * w[((oc * 3 + c) * 4 + dy) * 4 + dx];
  out[idx] = acc;
}

// ---------------------------------------------------------------------------
// LayerNorm over last dim C; one block per token.
// ---------------------------------------------------------------------------
__global__ void k_layernorm(const float* __restrict__ x,
                            const float* __restrict__ g,
                            const float* __restrict__ b,
                            float* __restrict__ y, int C) {
  __shared__ float part[8];
  __shared__ float s_mu, s_rstd;
  const int tok = blockIdx.x;
  const float* xr = x + (size_t)tok * C;
  float* yr = y + (size_t)tok * C;
  const int tid = threadIdx.x, lane = tid & 31, wv = tid >> 5;
  const int nwv = blockDim.x >> 5;

  float s = 0.f;
  for (int j = tid; j < C; j += blockDim.x) s += xr[j];
  for (int m = 16; m >= 1; m >>= 1) s += __shfl_xor(s, m, 32);
  if (lane == 0) part[wv] = s;
  __syncthreads();
  if (tid == 0) {
    float t = 0.f;
    for (int i = 0; i < nwv; ++i) t += part[i];
    s_mu = t / C;
  }
  __syncthreads();
  const float mu = s_mu;
  __syncthreads();

  float v = 0.f;
  for (int j = tid; j < C; j += blockDim.x) {
    float d = xr[j] - mu;
    v += d * d;
  }
  for (int m = 16; m >= 1; m >>= 1) v += __shfl_xor(v, m, 32);
  if (lane == 0) part[wv] = v;
  __syncthreads();
  if (tid == 0) {
    float t = 0.f;
    for (int i = 0; i < nwv; ++i) t += part[i];
    s_rstd = rsqrtf(t / C + 1e-5f);
  }
  __syncthreads();
  const float rstd = s_rstd;
  for (int j = tid; j < C; j += blockDim.x)
    yr[j] = (xr[j] - mu) * rstd * g[j] + b[j];
}

// ---------------------------------------------------------------------------
// Window partition (+cyclic shift) with bf16 convert.
// in: (B,H,W,C) fp32 -> out: (B*nH*nW, 49, C) bf16
// ---------------------------------------------------------------------------
__global__ void k_win_part(const float* __restrict__ x, bf16_t* __restrict__ out,
                           int Bn, int H, int W, int C, int shift) {
  size_t idx = (size_t)blockIdx.x * blockDim.x + threadIdx.x;
  size_t total = (size_t)Bn * H * W * C;
  if (idx >= total) return;
  int c = idx % C;         size_t t = idx / C;
  int i = t % 49;          t /= 49;
  int nW = W / 7;
  int wc = t % nW;         t /= nW;
  int nH = H / 7;
  int wr = t % nH;
  int bn = t / nH;
  int ih = i / 7, iw = i % 7;
  int gh = (wr * 7 + ih + shift) % H;
  int gw = (wc * 7 + iw + shift) % W;
  out[idx] = (bf16_t)x[(((size_t)bn * H + gh) * W + gw) * C + c];
}

// Window reverse (+unshift) with residual accumulate into x.
__global__ void k_win_rev_add(const float* __restrict__ win,
                              float* __restrict__ x,
                              int Bn, int H, int W, int C, int shift) {
  size_t idx = (size_t)blockIdx.x * blockDim.x + threadIdx.x;
  size_t total = (size_t)Bn * H * W * C;
  if (idx >= total) return;
  int c = idx % C;         size_t t = idx / C;
  int i = t % 49;          t /= 49;
  int nW = W / 7;
  int wc = t % nW;         t /= nW;
  int nH = H / 7;
  int wr = t % nH;
  int bn = t / nH;
  int ih = i / 7, iw = i % 7;
  int gh = (wr * 7 + ih + shift) % H;
  int gw = (wc * 7 + iw + shift) % W;
  x[(((size_t)bn * H + gh) * W + gw) * C + c] += win[idx];
}

// ---------------------------------------------------------------------------
// Generic GEMM: C[M,N] = A[M,K](bf16) @ W[N,K]^T(bf16) (+bias)(+resid)(+gelu)
// 128 threads = 4 waves; 128x64 block tile; wave w owns rows [32w,32w+32)
// (2 A-frags x 4 B-tiles = 8 WMMAs per K-step, B frag reused twice).
// Tiles staged to LDS with global_load_async_to_lds_b128 (ASYNCcnt).
// ---------------------------------------------------------------------------
__global__ __launch_bounds__(128)
void k_gemm(const bf16_t* __restrict__ A, const bf16_t* __restrict__ Bw,
            const float* __restrict__ bias, const float* resid, float* Cmat,
            int M, int N, int K, int gelu) {
  __shared__ __attribute__((aligned(64))) bf16_t As[128][32];
  __shared__ __attribute__((aligned(64))) bf16_t Bs[64][32];
  const int m0 = blockIdx.x * 128, n0 = blockIdx.y * 64;
  const int tid = threadIdx.x;
  const int wave = tid >> 5, lane = tid & 31;
  const int lhalf = lane >> 4, lmod = lane & 15;

  f32x8 acc[2][4] = {};

  const int arow_ld = tid;             // A: one 64B row per thread
  const int brow_ld = tid >> 1;        // B: half row (32B) per thread
  const int bcol_ld = (tid & 1) * 16;

  for (int k0 = 0; k0 < K; k0 += 32) {
    // ---- A tile: 128x32 bf16, async global->LDS; OOB rows zeroed ----
    {
      int gm = m0 + arow_ld;
      if (gm < M) {
        const bf16_t* src = A + (size_t)gm * K + k0;
        async_cp_row64((unsigned)(uintptr_t)&As[arow_ld][0],
                       (unsigned long long)(uintptr_t)src);
        if (k0 + 32 < K) __builtin_prefetch(src + 32, 0, 0);
      } else {
        uint4 z = make_uint4(0, 0, 0, 0);
        uint4* d = (uint4*)&As[arow_ld][0];
        d[0] = z; d[1] = z; d[2] = z; d[3] = z;
      }
    }
    // ---- B tile: 64x32 bf16 ----
    {
      int gn = n0 + brow_ld;
      if (gn < N) {
        const bf16_t* src = Bw + (size_t)gn * K + k0 + bcol_ld;
        async_cp_half32((unsigned)(uintptr_t)&Bs[brow_ld][bcol_ld],
                        (unsigned long long)(uintptr_t)src);
      } else {
        uint4 z = make_uint4(0, 0, 0, 0);
        uint4* d = (uint4*)&Bs[brow_ld][bcol_ld];
        d[0] = z; d[1] = z;
      }
    }
    wait_async0();        // own wave's async copies landed in LDS
    __syncthreads();      // ds_store zero-fill + all waves arrived

    Frag16 a[2];
#pragma unroll
    for (int sm = 0; sm < 2; ++sm) {
      const int ar = wave * 32 + sm * 16 + lmod;
      a[sm].u[0] = *(const uint4*)&As[ar][lhalf * 8];
      a[sm].u[1] = *(const uint4*)&As[ar][16 + lhalf * 8];
    }
#pragma unroll
    for (int nt = 0; nt < 4; ++nt) {
      Frag16 b;
      const int br = nt * 16 + lmod;
      b.u[0] = *(const uint4*)&Bs[br][lhalf * 8];
      b.u[1] = *(const uint4*)&Bs[br][16 + lhalf * 8];
      acc[0][nt] = wmma_bf16(a[0], b, acc[0][nt]);
      acc[1][nt] = wmma_bf16(a[1], b, acc[1][nt]);
    }
    __syncthreads();
  }

#pragma unroll
  for (int sm = 0; sm < 2; ++sm) {
#pragma unroll
    for (int nt = 0; nt < 4; ++nt) {
      int col = n0 + nt * 16 + lmod;
      if (col >= N) continue;
      float bv = bias ? bias[col] : 0.0f;
#pragma unroll
      for (int r = 0; r < 8; ++r) {
        int row = m0 + wave * 32 + sm * 16 + lhalf * 8 + r;
        if (row >= M) continue;
        float v = acc[sm][nt][r] + bv;
        if (resid) v += resid[(size_t)row * N + col];
        if (gelu)  v = 0.5f * v * (1.0f + erff(v * 0.7071067811865475f));
        Cmat[(size_t)row * N + col] = v;
      }
    }
  }
}

// ---------------------------------------------------------------------------
// Fused window attention: one block per (window, head). head_dim = 32 always.
// qkv: (nWin*49, 3C) fp32. out: (nWin*49, C) fp32.
// S = (Q*scale)K^T + rpb + shift-mask -> softmax -> O = P V, all via WMMA.
// ---------------------------------------------------------------------------
__global__ __launch_bounds__(128)
void k_win_attn(const float* __restrict__ qkv, const float* __restrict__ rpb,
                float* __restrict__ out, int C, int heads, int nWinPerImg,
                int nW, int shift, int H, int W) {
  __shared__ __attribute__((aligned(32))) bf16_t Qs[64][32];
  __shared__ __attribute__((aligned(32))) bf16_t Ks[64][32];
  __shared__ __attribute__((aligned(32))) bf16_t Vt[32][64];
  __shared__ __attribute__((aligned(32))) bf16_t Ps[64][64];
  const int win = blockIdx.x, h = blockIdx.y;
  const int tid = threadIdx.x;
  const int wave = tid >> 5, lane = tid & 31;
  const int lhalf = lane >> 4, lmod = lane & 15;
  const int C3 = 3 * C;
  const float scale = 0.17677669529663687f;  // 32^-0.5

  for (int idx = tid; idx < 64 * 32; idx += 128) {
    int tok = idx >> 5, d = idx & 31;
    float q = 0.f, k = 0.f, v = 0.f;
    if (tok < 49) {
      size_t base = (size_t)(win * 49 + tok) * C3 + h * 32 + d;
      q = qkv[base];
      k = qkv[base + C];
      v = qkv[base + 2 * C];
    }
    Qs[tok][d] = (bf16_t)(q * scale);
    Ks[tok][d] = (bf16_t)k;
    Vt[d][tok] = (bf16_t)v;
  }
  __syncthreads();

  // S = Q K^T : wave handles rows [16w, 16w+16), 4 col tiles.
  f32x8 s[4] = {};
  {
    Frag16 a;
    const int ar = wave * 16 + lmod;
    a.u[0] = *(const uint4*)&Qs[ar][lhalf * 8];
    a.u[1] = *(const uint4*)&Qs[ar][16 + lhalf * 8];
#pragma unroll
    for (int nt = 0; nt < 4; ++nt) {
      Frag16 b;
      const int br = nt * 16 + lmod;
      b.u[0] = *(const uint4*)&Ks[br][lhalf * 8];
      b.u[1] = *(const uint4*)&Ks[br][16 + lhalf * 8];
      s[nt] = wmma_bf16(a, b, s[nt]);
    }
  }

  // bias + mask + softmax (rows live in 16-lane half-groups of the D layout)
  const int wInImg = win % nWinPerImg;
  const int wr = wInImg / nW, wc = wInImg % nW;
#pragma unroll
  for (int r = 0; r < 8; ++r) {
    const int i = wave * 16 + lhalf * 8 + r;
    float vals[4];
    float vmax = -3.0e38f;
#pragma unroll
    for (int nt = 0; nt < 4; ++nt) {
      const int j = nt * 16 + lmod;
      float v;
      if (i < 49 && j < 49) {
        int ih = i / 7, iw = i % 7, jh = j / 7, jw = j % 7;
        int rel = (ih - jh + 6) * 13 + (iw - jw + 6);
        v = s[nt][r] + rpb[rel * heads + h];
        if (shift) {
          int ghi = wr * 7 + ih, gwi = wc * 7 + iw;
          int ghj = wr * 7 + jh, gwj = wc * 7 + jw;
          int ri = (ghi < H - 7) ? 0 : ((ghi < H - 3) ? 1 : 2);
          int rj = (ghj < H - 7) ? 0 : ((ghj < H - 3) ? 1 : 2);
          int ci = (gwi < W - 7) ? 0 : ((gwi < W - 3) ? 1 : 2);
          int cj = (gwj < W - 7) ? 0 : ((gwj < W - 3) ? 1 : 2);
          if (ri != rj || ci != cj) v -= 100.0f;
        }
      } else {
        v = -3.0e38f;
      }
      vals[nt] = v;
      vmax = fmaxf(vmax, v);
    }
    for (int m = 1; m < 16; m <<= 1) vmax = fmaxf(vmax, __shfl_xor(vmax, m, 32));
    float ssum = 0.f;
#pragma unroll
    for (int nt = 0; nt < 4; ++nt) {
      vals[nt] = __expf(vals[nt] - vmax);
      ssum += vals[nt];
    }
    for (int m = 1; m < 16; m <<= 1) ssum += __shfl_xor(ssum, m, 32);
    const float inv = 1.0f / ssum;
#pragma unroll
    for (int nt = 0; nt < 4; ++nt)
      Ps[i][nt * 16 + lmod] = (bf16_t)(vals[nt] * inv);
  }
  __syncthreads();

  // O = P V : rows [16w,16w+16) x 32 dims, K loop over 64 (2 steps).
  f32x8 o[2] = {};
  for (int kk = 0; kk < 64; kk += 32) {
    Frag16 a;
    const int ar = wave * 16 + lmod;
    a.u[0] = *(const uint4*)&Ps[ar][kk + lhalf * 8];
    a.u[1] = *(const uint4*)&Ps[ar][kk + 16 + lhalf * 8];
#pragma unroll
    for (int nt = 0; nt < 2; ++nt) {
      Frag16 b;
      const int d = nt * 16 + lmod;
      b.u[0] = *(const uint4*)&Vt[d][kk + lhalf * 8];
      b.u[1] = *(const uint4*)&Vt[d][kk + 16 + lhalf * 8];
      o[nt] = wmma_bf16(a, b, o[nt]);
    }
  }
#pragma unroll
  for (int nt = 0; nt < 2; ++nt) {
    const int d = nt * 16 + lmod;
#pragma unroll
    for (int r = 0; r < 8; ++r) {
      const int i = wave * 16 + lhalf * 8 + r;
      if (i < 49) out[(size_t)(win * 49 + i) * C + h * 32 + d] = o[nt][r];
    }
  }
}

// ---------------------------------------------------------------------------
// Patch-merge concat: (B,H,W,C) -> (B,H/2,W/2,4C), quadrants x0,x1,x2,x3.
// ---------------------------------------------------------------------------
__global__ void k_merge_cat(const float* __restrict__ x, float* __restrict__ out,
                            int Bn, int H, int W, int C) {
  size_t idx = (size_t)blockIdx.x * blockDim.x + threadIdx.x;
  size_t total = (size_t)Bn * (H / 2) * (W / 2) * 4 * C;
  if (idx >= total) return;
  int c4 = idx % (4 * C); size_t t = idx / (4 * C);
  int ow = t % (W / 2);   t /= (W / 2);
  int oh = t % (H / 2);
  int bn = t / (H / 2);
  int q = c4 / C, c = c4 % C;
  int dy = (q == 1 || q == 3) ? 1 : 0;
  int dx = (q >= 2) ? 1 : 0;
  out[idx] = x[(((size_t)bn * H + 2 * oh + dy) * W + 2 * ow + dx) * C + c];
}

// Mean over tokens per batch.
__global__ void k_meanpool(const float* __restrict__ x, float* __restrict__ out,
                           int ntokPerB, int C) {
  int bn = blockIdx.x;
  for (int c = threadIdx.x; c < C; c += blockDim.x) {
    float s = 0.f;
    for (int t = 0; t < ntokPerB; ++t)
      s += x[((size_t)bn * ntokPerB + t) * C + c];
    out[bn * C + c] = s / ntokPerB;
  }
}

// Head: (16,768)@(4,768)^T + b -> logits; write tuple (logits, logits).
__global__ void k_head(const float* __restrict__ x, const float* __restrict__ w,
                       const float* __restrict__ b, float* __restrict__ out) {
  int idx = threadIdx.x;
  if (idx >= 64) return;
  int bn = idx >> 2, n = idx & 3;
  float s = b[n];
  for (int k = 0; k < 768; ++k) s += x[bn * 768 + k] * w[n * 768 + k];
  out[idx] = s;
  out[64 + idx] = s;
}

// ---------------------------------------------------------------------------
// Driver
// ---------------------------------------------------------------------------
extern "C" void kernel_launch(void* const* d_in, const int* in_sizes, int n_in,
                              void* d_out, int out_size, void* d_ws,
                              size_t ws_size, hipStream_t stream) {
  (void)in_sizes; (void)n_in; (void)out_size; (void)ws_size;

  // --- input mapping: x first, then params leaves in sorted pytree order ---
  struct Blk {
    const float *fc1_b, *fc1_w, *fc2_b, *fc2_w, *ln1_b, *ln1_g, *ln2_b, *ln2_g,
                *proj_b, *proj_w, *qkv_b, *qkv_w, *rpb;
  } blk[12];
  struct Mrg { const float *ln_b, *ln_g, *red_w; } mrg[3];

  int p = 0;
  const float* x_in = (const float*)d_in[p++];
  for (int i = 0; i < 12; ++i) {
    blk[i].fc1_b  = (const float*)d_in[p++];
    blk[i].fc1_w  = (const float*)d_in[p++];
    blk[i].fc2_b  = (const float*)d_in[p++];
    blk[i].fc2_w  = (const float*)d_in[p++];
    blk[i].ln1_b  = (const float*)d_in[p++];
    blk[i].ln1_g  = (const float*)d_in[p++];
    blk[i].ln2_b  = (const float*)d_in[p++];
    blk[i].ln2_g  = (const float*)d_in[p++];
    blk[i].proj_b = (const float*)d_in[p++];
    blk[i].proj_w = (const float*)d_in[p++];
    blk[i].qkv_b  = (const float*)d_in[p++];
    blk[i].qkv_w  = (const float*)d_in[p++];
    blk[i].rpb    = (const float*)d_in[p++];
  }
  const float* head_b = (const float*)d_in[p++];
  const float* head_w = (const float*)d_in[p++];
  const float* ln0_b  = (const float*)d_in[p++];
  const float* ln0_g  = (const float*)d_in[p++];
  const float* lnf_b  = (const float*)d_in[p++];
  const float* lnf_g  = (const float*)d_in[p++];
  for (int i = 0; i < 3; ++i) {
    mrg[i].ln_b  = (const float*)d_in[p++];
    mrg[i].ln_g  = (const float*)d_in[p++];
    mrg[i].red_w = (const float*)d_in[p++];
  }
  const float* patch_b = (const float*)d_in[p++];
  const float* patch_w = (const float*)d_in[p++];

  // --- workspace carve (fixed byte offsets) ---
  char* wsb = (char*)d_ws;
  const size_t MB = 1024 * 1024;
  float*  X    = (float*)(wsb + 0 * MB);     // residual stream  (<=19.3MB)
  float*  LNb  = (float*)(wsb + 20 * MB);    // LN out / attn out (<=19.3MB)
  float*  CAT  = (float*)(wsb + 40 * MB);    // merge concat / proj out
  float*  G    = (float*)(wsb + 60 * MB);    // gemm out (qkv / fc1) (<=78MB)
  bf16_t* Abf  = (bf16_t*)(wsb + 140 * MB);  // bf16 activations (<=39MB)
  bf16_t* Wbf  = (bf16_t*)(wsb + 180 * MB);  // bf16 weights (<=5MB)
  float*  POOL = (float*)(wsb + 186 * MB);   // (16,768)

  auto gemm = [&](const bf16_t* A, const bf16_t* Bw, const float* bias,
                  const float* resid, float* out, int M, int N, int K,
                  int gelu) {
    dim3 g((M + 127) / 128, (N + 63) / 64);
    k_gemm<<<g, 128, 0, stream>>>(A, Bw, bias, resid, out, M, N, K, gelu);
  };
  auto tobf = [&](const float* in, bf16_t* out, size_t n) {
    k_f2bf<<<(unsigned)((n + 255) / 256), 256, 0, stream>>>(in, out, n);
  };
  auto ln = [&](const float* in, const float* g, const float* b, float* out,
                int ntok, int C) {
    k_layernorm<<<ntok, 256, 0, stream>>>(in, g, b, out, C);
  };

  // --- patch embed + initial LN ---
  {
    int total = 16 * 56 * 56 * 96;
    k_patch_embed<<<(total + 255) / 256, 256, 0, stream>>>(x_in, patch_w,
                                                           patch_b, CAT);
    ln(CAT, ln0_g, ln0_b, X, 16 * 56 * 56, 96);
  }

  static const int DEP[4]   = {2, 2, 6, 2};
  static const int HEADS[4] = {3, 6, 12, 24};
  int H = 56, W = 56, C = 96;
  int bi = 0;
  for (int s = 0; s < 4; ++s) {
    const int heads = HEADS[s];
    for (int d = 0; d < DEP[s]; ++d, ++bi) {
      const Blk& B = blk[bi];
      const int shift = (d & 1) ? 3 : 0;
      const int ntok = 16 * H * W;
      const int nH = H / 7, nW = W / 7;
      const int nWin = 16 * nH * nW;
      const size_t elems = (size_t)ntok * C;

      // ---- window attention ----
      ln(X, B.ln1_g, B.ln1_b, LNb, ntok, C);
      k_win_part<<<(unsigned)((elems + 255) / 256), 256, 0, stream>>>(
          LNb, Abf, 16, H, W, C, shift);
      tobf(B.qkv_w, Wbf, (size_t)3 * C * C);
      gemm(Abf, Wbf, B.qkv_b, nullptr, G, ntok, 3 * C, C, 0);
      {
        dim3 g(nWin, heads);
        k_win_attn<<<g, 128, 0, stream>>>(G, B.rpb, LNb, C, heads, nH * nW,
                                          nW, shift, H, W);
      }
      tobf(LNb, Abf, elems);
      tobf(B.proj_w, Wbf, (size_t)C * C);
      gemm(Abf, Wbf, B.proj_b, nullptr, CAT, ntok, C, C, 0);
      k_win_rev_add<<<(unsigned)((elems + 255) / 256), 256, 0, stream>>>(
          CAT, X, 16, H, W, C, shift);

      // ---- MLP ----
      ln(X, B.ln2_g, B.ln2_b, LNb, ntok, C);
      tobf(LNb, Abf, elems);
      tobf(B.fc1_w, Wbf, (size_t)4 * C * C);
      gemm(Abf, Wbf, B.fc1_b, nullptr, G, ntok, 4 * C, C, 1);  // +GELU
      tobf(G, Abf, (size_t)ntok * 4 * C);
      tobf(B.fc2_w, Wbf, (size_t)4 * C * C);
      gemm(Abf, Wbf, B.fc2_b, X, X, ntok, C, 4 * C, 0);  // +residual, in place
    }
    if (s < 3) {
      const Mrg& M2 = mrg[s];
      const int ntok2 = 16 * (H / 2) * (W / 2);
      const size_t el2 = (size_t)ntok2 * 4 * C;
      k_merge_cat<<<(unsigned)((el2 + 255) / 256), 256, 0, stream>>>(
          X, CAT, 16, H, W, C);
      ln(CAT, M2.ln_g, M2.ln_b, LNb, ntok2, 4 * C);
      tobf(LNb, Abf, el2);
      tobf(M2.red_w, Wbf, (size_t)2 * C * 4 * C);
      gemm(Abf, Wbf, nullptr, nullptr, X, ntok2, 2 * C, 4 * C, 0);
      H >>= 1; W >>= 1; C <<= 1;
    }
  }

  // --- final LN, mean pool, head ---
  ln(X, lnf_g, lnf_b, LNb, 16 * 7 * 7, 768);
  k_meanpool<<<16, 256, 0, stream>>>(LNb, POOL, 49, 768);
  k_head<<<1, 64, 0, stream>>>(POOL, head_w, head_b, (float*)d_out);
}
